// Loss_4861902979528
// MI455X (gfx1250) — matrix operand order repaired and checked by reading, not compiled, not run on hardware
//
#include <hip/hip_runtime.h>
#include <hip/hip_bf16.h>

#define N_ROWS 12288
#define D_DIM  128
#define KSPLIT 8
#define KCHUNK (N_ROWS / KSPLIT)   // 1536
#define LDS_STRIDE 34              // 32 + 2 pad (ushorts) -> conflict-free frag reads

typedef __attribute__((ext_vector_type(16))) __bf16 v16bf;
typedef __attribute__((ext_vector_type(8)))  float  v8f;

__device__ __forceinline__ unsigned short f2bf(float f) {
  unsigned int u = __float_as_uint(f);
  u += 0x7FFFu + ((u >> 16) & 1u);          // round-to-nearest-even
  return (unsigned short)(u >> 16);
}

union AFrag { v16bf v; unsigned int w[8]; };
union BFrag { v16bf v; unsigned int w[8]; };

// ---------------------------------------------------------------------------
// Phase A: L4-normalize rows, emit e (f32), log_p transposed (bf16), p_log_p.
// Also zero the RS/RA/U accumulators.
// One wave per row; 8 rows per 256-thread block.
// ---------------------------------------------------------------------------
__global__ __launch_bounds__(256) void prep_kernel(
    const float* __restrict__ emb,
    float* __restrict__ ebuf,             // [N, 128] f32
    unsigned short* __restrict__ lpT,     // [128, N] bf16 (transposed)
    float* __restrict__ plp,              // [N]
    float* __restrict__ RS, float* __restrict__ RA, float* __restrict__ U) {
  const int tid = threadIdx.x;
  const int wv  = tid >> 5;
  const int ln  = tid & 31;
  const int row = blockIdx.x * 8 + wv;

  if (tid < 8) {
    const int r = blockIdx.x * 8 + tid;
    RS[r] = 0.0f; RA[r] = 0.0f; U[r] = 0.0f;
  }

  float x[4];
  float s4 = 0.0f;
#pragma unroll
  for (int q = 0; q < 4; ++q) {
    x[q] = emb[(size_t)row * D_DIM + ln + 32 * q];
    const float x2 = x[q] * x[q];
    s4 += x2 * x2;                        // |x|^4
  }
#pragma unroll
  for (int m = 16; m >= 1; m >>= 1) s4 += __shfl_xor(s4, m);

  const float norm4 = sqrtf(sqrtf(s4));
  const float inv   = 1.0f / fmaxf(norm4, 1e-12f);

  float pl = 0.0f;
#pragma unroll
  for (int q = 0; q < 4; ++q) {
    const int d = ln + 32 * q;
    const float e  = x[q] * inv;
    const float lp = logf(e + 1e-10f);
    pl += e * lp;
    ebuf[(size_t)row * D_DIM + d] = e;
    lpT[(size_t)d * N_ROWS + row] = f2bf(lp);
  }
#pragma unroll
  for (int m = 16; m >= 1; m >>= 1) pl += __shfl_xor(pl, m);
  if (ln == 0) plp[row] = pl;
}

// ---------------------------------------------------------------------------
// Phase B: Q = adj @ log_p via v_wmma_f32_16x16x32_bf16, fused with per-row
// adj sums (RS/RA) and the e-weighted reduction U[i] = sum_d Q[i,d]*e[i,d].
// Block: 256 threads = 8 waves. Each wave: 16 rows x 128 cols output strip.
// Grid: (N/128 row blocks, KSPLIT k-chunks). adj is read exactly once.
// ---------------------------------------------------------------------------
__global__ __launch_bounds__(256) void gemm_kl_kernel(
    const float* __restrict__ adj,
    const unsigned short* __restrict__ lpT,
    const float* __restrict__ ebuf,
    float* __restrict__ RS, float* __restrict__ RA, float* __restrict__ U) {
  __shared__ __align__(16) unsigned short As[128 * LDS_STRIDE];

  const int tid = threadIdx.x;
  const int wv  = tid >> 5;           // wave 0..7 -> row strip
  const int ln  = tid & 31;
  const int hi  = ln >> 4;            // half-wave select
  const int l16 = ln & 15;

  const int rb = blockIdx.x * 128;        // block row base
  const int k0base = blockIdx.y * KCHUNK; // k window base

  // A staging assignment: thread covers rows {p*32 + (tid>>3)} , float4 col tid&7
  const int arow = tid >> 3;          // 0..31
  const int ac4  = tid & 7;           // 0..7

  float rs[4] = {0.f, 0.f, 0.f, 0.f};
  float ra[4] = {0.f, 0.f, 0.f, 0.f};

  v8f acc[8] = {};                    // 16x128 f32 partial Q tile

  const int afrag_row = wv * 16 + l16;

  for (int kk = 0; kk < KCHUNK; kk += 32) {
    const int k0 = k0base + kk;

    // ---- coalesced global read of adj[rb..rb+127][k0..k0+31] ----
    float4 ld[4];
#pragma unroll
    for (int p = 0; p < 4; ++p) {
      const int r = p * 32 + arow;
      ld[p] = *reinterpret_cast<const float4*>(
          adj + (size_t)(rb + r) * N_ROWS + k0 + ac4 * 4);
      rs[p] += ld[p].x + ld[p].y + ld[p].z + ld[p].w;
      ra[p] += fabsf(ld[p].x) + fabsf(ld[p].y) + fabsf(ld[p].z) + fabsf(ld[p].w);
    }
    __syncthreads();
#pragma unroll
    for (int p = 0; p < 4; ++p) {
      const int r = p * 32 + arow;
      unsigned short* dst = &As[r * LDS_STRIDE + ac4 * 4];
      dst[0] = f2bf(ld[p].x); dst[1] = f2bf(ld[p].y);
      dst[2] = f2bf(ld[p].z); dst[3] = f2bf(ld[p].w);
    }
    __syncthreads();

    // ---- A fragment (16x32 bf16, contiguous k order) from LDS ----
    AFrag a;
#pragma unroll
    for (int v = 0; v < 8; ++v) {
      const int kv = hi * 16 + 2 * v;  // even
      a.w[v] = *reinterpret_cast<const unsigned int*>(
          &As[afrag_row * LDS_STRIDE + kv]);
    }

    // ---- 8 column tiles: B direct from transposed log_p (L2-resident) ----
#pragma unroll
    for (int t = 0; t < 8; ++t) {
      const int d = t * 16 + l16;     // B column = lane
      const v16bf b = *reinterpret_cast<const v16bf*>(
          lpT + (size_t)d * N_ROWS + k0 + hi * 16);   // 32B aligned
      acc[t] = __builtin_amdgcn_wmma_f32_16x16x32_bf16(
          false, a.v, false, b, (short)0, acc[t], false, false);
    }
  }

  // ---- fused row sums of adj ----
#pragma unroll
  for (int p = 0; p < 4; ++p) {
    atomicAdd(&RS[rb + p * 32 + arow], rs[p]);
    atomicAdd(&RA[rb + p * 32 + arow], ra[p]);
  }

  // ---- epilogue: U[i] += sum_d Q[i,d] * e[i,d] (C layout: VGPR r -> M=r+8*hi)
#pragma unroll
  for (int r = 0; r < 8; ++r) {
    const int grow = rb + wv * 16 + hi * 8 + r;
    float us = 0.0f;
#pragma unroll
    for (int t = 0; t < 8; ++t) {
      const int gcol = t * 16 + l16;
      us += acc[t][r] * ebuf[(size_t)grow * D_DIM + gcol];
    }
#pragma unroll
    for (int m = 8; m >= 1; m >>= 1) us += __shfl_xor(us, m);
    if (l16 == 0) atomicAdd(&U[grow], us);
  }
}

// ---------------------------------------------------------------------------
// Phase C: scalar loss = lambda * sum_i (plp[i]*RS[i] - U[i]) / max(RA[i],eps)
// ---------------------------------------------------------------------------
__global__ __launch_bounds__(256) void finalize_kernel(
    const float* __restrict__ plp, const float* __restrict__ RS,
    const float* __restrict__ RA, const float* __restrict__ U,
    const float* __restrict__ lam, float* __restrict__ out) {
  __shared__ float sh[8];
  float a = 0.0f;
  for (int i = threadIdx.x; i < N_ROWS; i += 256)
    a += (plp[i] * RS[i] - U[i]) / fmaxf(RA[i], 1e-12f);
#pragma unroll
  for (int m = 16; m >= 1; m >>= 1) a += __shfl_xor(a, m);
  if ((threadIdx.x & 31) == 0) sh[threadIdx.x >> 5] = a;
  __syncthreads();
  if (threadIdx.x == 0) {
    float t = 0.0f;
#pragma unroll
    for (int i = 0; i < 8; ++i) t += sh[i];
    out[0] = lam[0] * t;
  }
}

// ---------------------------------------------------------------------------
extern "C" void kernel_launch(void* const* d_in, const int* in_sizes, int n_in,
                              void* d_out, int out_size, void* d_ws, size_t ws_size,
                              hipStream_t stream) {
  (void)in_sizes; (void)n_in; (void)out_size; (void)ws_size;
  const float* emb = (const float*)d_in[0];
  const float* adj = (const float*)d_in[1];
  const float* lam = (const float*)d_in[2];
  float* out = (float*)d_out;

  char* ws = (char*)d_ws;
  float*          ebuf = (float*)(ws + 0);                       // 6,291,456 B
  unsigned short* lpT  = (unsigned short*)(ws + 6291456);        // 3,145,728 B
  float*          plp  = (float*)(ws + 9437184);                 //    49,152 B
  float*          RS   = (float*)(ws + 9486336);
  float*          RA   = (float*)(ws + 9535488);
  float*          U    = (float*)(ws + 9584640);

  prep_kernel<<<N_ROWS / 8, 256, 0, stream>>>(emb, ebuf, lpT, plp, RS, RA, U);
  gemm_kl_kernel<<<dim3(N_ROWS / 128, KSPLIT), 256, 0, stream>>>(
      adj, lpT, ebuf, RS, RA, U);
  finalize_kernel<<<1, 256, 0, stream>>>(plp, RS, RA, U, lam, out);
}
